// LSTMModel_40638980555081
// MI455X (gfx1250) — matrix operand order repaired and checked by reading, not compiled, not run on hardware
//
#include <hip/hip_runtime.h>

#define S_LEN 512
#define BATCHN 128
#define EMBN 256
#define HIDN 512
#define KTOT 768
#define WROW 776          // padded LDS row (elements) to avoid bank conflicts
#define NWG 64
#define HBUF (BATCHN*HIDN)  // 65536 elements per h buffer

typedef __attribute__((ext_vector_type(16))) __bf16 v16bf;
typedef __attribute__((ext_vector_type(8)))  float  v8f;
typedef __attribute__((ext_vector_type(4)))  unsigned int v4u;
typedef __attribute__((ext_vector_type(4)))  float v4f;

union Frag { v16bf v; v4u q[2]; };

__device__ __forceinline__ unsigned short f2bf(float f) {
  unsigned int x = __float_as_uint(f);
  unsigned int r = x + 0x7FFFu + ((x >> 16) & 1u);   // round-to-nearest-even
  return (unsigned short)(r >> 16);
}
__device__ __forceinline__ float sigmoid_f(float x) { return 1.0f / (1.0f + __expf(-x)); }
__device__ __forceinline__ float tanh_f(float x) {
  float e = __expf(2.0f * x);
  return 1.0f - 2.0f / (e + 1.0f);
}

// ---------------- Phase 0a: weight pack (bf16 [2048][768]), bias sum, zero-init ----
extern "C" __global__ void prep_kernel(const float* __restrict__ Wih,
                                       const float* __restrict__ Whh,
                                       const float* __restrict__ bih,
                                       const float* __restrict__ bhh,
                                       const float* __restrict__ bfc,
                                       unsigned short* __restrict__ Wbf,
                                       float* __restrict__ bsum,
                                       unsigned short* __restrict__ hbf,
                                       float* __restrict__ out,
                                       unsigned int* __restrict__ cnt) {
  int idx = blockIdx.x * blockDim.x + threadIdx.x;      // exactly 2048*768 threads
  int g = idx / KTOT;
  int c = idx - g * KTOT;
  float v = (c < EMBN) ? Wih[g * EMBN + c] : Whh[g * HIDN + (c - EMBN)];
  Wbf[g * KTOT + c] = f2bf(v);
  if (idx < 2048)   bsum[idx] = bih[idx] + bhh[idx];
  if (idx < 2*HBUF) hbf[idx] = 0;                       // h0 = 0 (both ping-pong bufs)
  if (idx < BATCHN * 2) out[idx] = bfc[idx & 1];        // out init = b_fc
  if (idx == 0) *cnt = 0u;
}

// ---------------- Phase 0b: embedding gather -> bf16 X[s][b][e] -------------------
extern "C" __global__ void embed_kernel(const int* __restrict__ text,
                                        const float* __restrict__ emb,
                                        unsigned short* __restrict__ Xbf) {
  int idx = blockIdx.x * blockDim.x + threadIdx.x;      // 2,097,152 threads x 8 elems
  int base = idx << 3;
  int e = base & (EMBN - 1);
  int b = (base >> 8) & (BATCHN - 1);
  int s = base >> 15;
  int tok = text[s * BATCHN + b];
  const float* src = emb + (size_t)tok * EMBN + e;
  v4f f0 = *(const v4f*)src;
  v4f f1 = *(const v4f*)(src + 4);
  v4u o;
  o.x = (unsigned)f2bf(f0.x) | ((unsigned)f2bf(f0.y) << 16);
  o.y = (unsigned)f2bf(f0.z) | ((unsigned)f2bf(f0.w) << 16);
  o.z = (unsigned)f2bf(f1.x) | ((unsigned)f2bf(f1.y) << 16);
  o.w = (unsigned)f2bf(f1.z) | ((unsigned)f2bf(f1.w) << 16);
  *(v4u*)(Xbf + base) = o;
}

// ---------------- Phase 1: persistent sequential LSTM (WMMA bf16->f32) ------------
// Grid: 64 WGs = 32 hidden slices x 2 batch halves, 128 threads (4 waves) each.
// Wave owns batch tile [batchBase..+16) x hidden tile [hb..+16) for all 4 gates.
extern "C" __global__ __launch_bounds__(128, 1)
void lstm_seq_kernel(const unsigned short* __restrict__ Xbf,   // [512][128][256] bf16
                     const unsigned short* __restrict__ Wbf,   // [2048][768] bf16
                     const float* __restrict__ bsum,           // [2048]
                     unsigned short* __restrict__ hbf,         // [2][128][512] bf16
                     const float* __restrict__ Wfc,            // [2][512] f32
                     float* __restrict__ out,                  // [128][2] f32
                     unsigned int* __restrict__ cnt) {
  extern __shared__ unsigned short Wlds[];                     // [64][WROW], LDS offset 0

  const int tid  = threadIdx.x;
  const int lane = tid & 31;
  const int wave = tid >> 5;
  const int hidx = blockIdx.x & 31;
  const int hb   = hidx << 4;                                  // hidden base
  const int bhalf = blockIdx.x >> 5;
  const int batchBase = bhalf * 64 + wave * 16;

  // ---- async-copy this WG's 64 gate rows (i,f,g,o for 16 hidden units) into LDS.
  // Thread t owns half of gate-row (t>>1); 48 x 16B async chunks per thread.
  {
    const int r    = tid >> 1;                 // 0..63  (local gate row)
    const int rhal = tid & 1;                  // which 768B half of the row
    const int t    = r >> 4, rr = r & 15;
    const unsigned short* gsrc =
        Wbf + (size_t)(t * HIDN + hb + rr) * KTOT + rhal * 384;
    unsigned int ldsoff = (unsigned int)(r * WROW + rhal * 384) * 2u;  // byte offset
    #pragma unroll
    for (int j = 0; j < 48; ++j) {
      asm volatile("global_load_async_to_lds_b128 %0, %1, off"
                   :: "v"(ldsoff + (unsigned)j * 16u), "v"(gsrc + j * 8)
                   : "memory");
    }
    asm volatile("s_wait_asynccnt 0x0" ::: "memory");
  }
  __syncthreads();

  const int colh  = lane & 15;       // hidden col in tile; also batch row for A frag
  const int half  = lane >> 4;
  const int rowA  = batchBase + colh;
  const int khalf = half * 16;       // A chunk base within a K-tile of 32

  float bias[4];
  #pragma unroll
  for (int t = 0; t < 4; ++t) bias[t] = bsum[t * HIDN + hb + colh];
  const float w0 = Wfc[hb + colh];
  const float w1 = Wfc[HIDN + hb + colh];

  const unsigned short* bbase = Wlds + colh * WROW + half * 16;  // gate-type stride 16*WROW

  v8f cstate = (v8f)0.0f;            // cell state lives in VGPRs for all 512 steps

  for (int s = 0; s < S_LEN; ++s) {
    const unsigned short* hcur = hbf + ((s & 1) ? HBUF : 0);
    unsigned short* hnext      = hbf + ((s & 1) ? 0 : HBUF);

    v8f acc[4];
    #pragma unroll
    for (int t = 0; t < 4; ++t) acc[t] = (v8f)bias[t];

    // Warm L2/WGP$ for next step's x_t row while this step computes.
    __builtin_prefetch(Xbf + (((size_t)(s + 1) << 7) + rowA) * EMBN + khalf, 0, 1);

    // K tiles 0..7 from x_t (E=256)
    const unsigned short* xrow = Xbf + (((size_t)s << 7) + rowA) * EMBN + khalf;
    #pragma unroll
    for (int kt = 0; kt < 8; ++kt) {
      Frag a;
      a.q[0] = *(const v4u*)(xrow + kt * 32);
      a.q[1] = *(const v4u*)(xrow + kt * 32 + 8);
      Frag b0, b1, b2, b3;
      const unsigned short* bp = bbase + kt * 32;
      b0.q[0] = *(const v4u*)(bp);              b0.q[1] = *(const v4u*)(bp + 8);
      b1.q[0] = *(const v4u*)(bp + 16*WROW);    b1.q[1] = *(const v4u*)(bp + 16*WROW + 8);
      b2.q[0] = *(const v4u*)(bp + 32*WROW);    b2.q[1] = *(const v4u*)(bp + 32*WROW + 8);
      b3.q[0] = *(const v4u*)(bp + 48*WROW);    b3.q[1] = *(const v4u*)(bp + 48*WROW + 8);
      acc[0] = __builtin_amdgcn_wmma_f32_16x16x32_bf16(false, a.v, false, b0.v, (short)0, acc[0], false, false);
      acc[1] = __builtin_amdgcn_wmma_f32_16x16x32_bf16(false, a.v, false, b1.v, (short)0, acc[1], false, false);
      acc[2] = __builtin_amdgcn_wmma_f32_16x16x32_bf16(false, a.v, false, b2.v, (short)0, acc[2], false, false);
      acc[3] = __builtin_amdgcn_wmma_f32_16x16x32_bf16(false, a.v, false, b3.v, (short)0, acc[3], false, false);
    }
    // K tiles 8..23 from h_{t-1} (H=512)
    const unsigned short* hrow = hcur + rowA * HIDN + khalf;
    #pragma unroll
    for (int kt = 0; kt < 16; ++kt) {
      Frag a;
      a.q[0] = *(const v4u*)(hrow + kt * 32);
      a.q[1] = *(const v4u*)(hrow + kt * 32 + 8);
      Frag b0, b1, b2, b3;
      const unsigned short* bp = bbase + (kt + 8) * 32;
      b0.q[0] = *(const v4u*)(bp);              b0.q[1] = *(const v4u*)(bp + 8);
      b1.q[0] = *(const v4u*)(bp + 16*WROW);    b1.q[1] = *(const v4u*)(bp + 16*WROW + 8);
      b2.q[0] = *(const v4u*)(bp + 32*WROW);    b2.q[1] = *(const v4u*)(bp + 32*WROW + 8);
      b3.q[0] = *(const v4u*)(bp + 48*WROW);    b3.q[1] = *(const v4u*)(bp + 48*WROW + 8);
      acc[0] = __builtin_amdgcn_wmma_f32_16x16x32_bf16(false, a.v, false, b0.v, (short)0, acc[0], false, false);
      acc[1] = __builtin_amdgcn_wmma_f32_16x16x32_bf16(false, a.v, false, b1.v, (short)0, acc[1], false, false);
      acc[2] = __builtin_amdgcn_wmma_f32_16x16x32_bf16(false, a.v, false, b2.v, (short)0, acc[2], false, false);
      acc[3] = __builtin_amdgcn_wmma_f32_16x16x32_bf16(false, a.v, false, b3.v, (short)0, acc[3], false, false);
    }

    // ---- elementwise LSTM cell (PyTorch gate order i,f,g,o) ----
    v8f hval;
    #pragma unroll
    for (int v = 0; v < 8; ++v) {
      float ii = sigmoid_f(acc[0][v]);
      float ff = sigmoid_f(acc[1][v]);
      float gg = tanh_f(acc[2][v]);
      float oo = sigmoid_f(acc[3][v]);
      float cc = ff * cstate[v] + ii * gg;
      cstate[v] = cc;
      hval[v] = oo * tanh_f(cc);
    }

    // ---- write h_new (bf16) for next step ----
    #pragma unroll
    for (int v = 0; v < 8; ++v) {
      int row = batchBase + (half << 3) + v;
      hnext[row * HIDN + hb + colh] = f2bf(hval[v]);
    }

    // ---- last step: fold in the FC head (OUT=2) via shuffle reduce + atomics ----
    if (s == S_LEN - 1) {
      #pragma unroll
      for (int v = 0; v < 8; ++v) {
        float p0 = hval[v] * w0;
        float p1 = hval[v] * w1;
        #pragma unroll
        for (int mk = 8; mk >= 1; mk >>= 1) {
          p0 += __shfl_xor(p0, mk, 32);
          p1 += __shfl_xor(p1, mk, 32);
        }
        if (colh == 0) {
          int row = batchBase + (half << 3) + v;
          atomicAdd(&out[row * 2 + 0], p0);
          atomicAdd(&out[row * 2 + 1], p1);
        }
      }
    }

    // ---- device-wide barrier: release h_new, acquire everyone else's ----
    __threadfence();
    __syncthreads();
    if (tid == 0) {
      __hip_atomic_fetch_add(cnt, 1u, __ATOMIC_ACQ_REL, __HIP_MEMORY_SCOPE_AGENT);
      unsigned int target = (unsigned int)(s + 1) * NWG;
      while (__hip_atomic_load(cnt, __ATOMIC_ACQUIRE, __HIP_MEMORY_SCOPE_AGENT) < target) {
        __builtin_amdgcn_s_sleep(1);
      }
    }
    __syncthreads();
  }
}

extern "C" void kernel_launch(void* const* d_in, const int* in_sizes, int n_in,
                              void* d_out, int out_size, void* d_ws, size_t ws_size,
                              hipStream_t stream) {
  const int*   text = (const int*)d_in[0];
  const float* emb  = (const float*)d_in[1];
  const float* Wih  = (const float*)d_in[2];
  const float* Whh  = (const float*)d_in[3];
  const float* bih  = (const float*)d_in[4];
  const float* bhh  = (const float*)d_in[5];
  const float* Wfc  = (const float*)d_in[6];
  const float* bfc  = (const float*)d_in[7];
  float* out = (float*)d_out;

  char* ws = (char*)d_ws;
  unsigned short* Xbf = (unsigned short*)ws; ws += (size_t)S_LEN * BATCHN * EMBN * 2;  // 32 MB
  unsigned short* Wbf = (unsigned short*)ws; ws += (size_t)2048 * KTOT * 2;            // 3 MB
  float*          bs  = (float*)ws;          ws += 2048 * 4;
  unsigned short* hbf = (unsigned short*)ws; ws += (size_t)2 * HBUF * 2;               // 256 KB
  unsigned int*   cnt = (unsigned int*)ws;

  // Phase 0: pack weights/bias, zero h, init out with b_fc, reset barrier counter
  prep_kernel<<<6144, 256, 0, stream>>>(Wih, Whh, bih, bhh, bfc, Wbf, bs, hbf, out, cnt);
  // Phase 0b: embedding gather to bf16
  embed_kernel<<<8192, 256, 0, stream>>>(text, emb, Xbf);
  // Phase 1: persistent sequential LSTM + FC head
  lstm_seq_kernel<<<NWG, 128, 64 * WROW * sizeof(unsigned short), stream>>>(
      Xbf, Wbf, bs, hbf, Wfc, out, cnt);
}
